// NERLSTM_CRF_64725157150772
// MI455X (gfx1250) — matrix operand order repaired and checked by reading, not compiled, not run on hardware
//
#include <hip/hip_runtime.h>
#include <hip/hip_bf16.h>
#include <math.h>

#define MAXLEN 70
#define NB     512          // batch
#define WD     300
#define FN     100
#define HDIM   256
#define NT     9
#define EDIM   16
#define DIN    716
#define DP     736          // DIN padded to 23*32 for K of WMMA GEMM
#define G4H    1024         // 4*H
#define MROWS  (MAXLEN*NB)  // 35840

typedef __attribute__((ext_vector_type(16))) __bf16 bf16x16;
typedef __attribute__((ext_vector_type(8)))  __bf16 bf16x8;
typedef __attribute__((ext_vector_type(8)))  float  f32x8;

__device__ __forceinline__ __bf16 f2bf(float f) {
  unsigned u = __builtin_bit_cast(unsigned, f);
  unsigned r = (u + 0x7FFFu + ((u >> 16) & 1u)) >> 16;   // RNE
  return __builtin_bit_cast(__bf16, (unsigned short)r);
}
__device__ __forceinline__ float bf2f(__bf16 h) {
  unsigned short s = __builtin_bit_cast(unsigned short, h);
  unsigned u = ((unsigned)s) << 16;
  return __builtin_bit_cast(float, u);
}
__device__ __forceinline__ float sigm(float x) { return 1.0f / (1.0f + expf(-x)); }

#define SHUF16(lo, hi) __builtin_shufflevector(lo, hi, 0,1,2,3,4,5,6,7,8,9,10,11,12,13,14,15)

// ---------------------------------------------------------------------------
// Weight conversion / padding to bf16, bias combine
// ---------------------------------------------------------------------------
__global__ __launch_bounds__(256) void convert_weights(
    const float* __restrict__ wihf, const float* __restrict__ wihb,
    const float* __restrict__ whhf, const float* __restrict__ whhb,
    const float* __restrict__ bihf, const float* __restrict__ bhhf,
    const float* __restrict__ bihb, const float* __restrict__ bhhb,
    __bf16* __restrict__ wihf16, __bf16* __restrict__ wihb16,
    __bf16* __restrict__ whhf16, __bf16* __restrict__ whhb16,
    float* __restrict__ biasf, float* __restrict__ biasb)
{
  long idx = (long)blockIdx.x * 256 + threadIdx.x;
  const long NW = (long)G4H * DP;
  const long NH = (long)G4H * HDIM;
  if (idx < NW) { long n = idx / DP, k = idx % DP;
    wihf16[idx] = f2bf(k < DIN ? wihf[n * DIN + k] : 0.0f); return; }
  idx -= NW;
  if (idx < NW) { long n = idx / DP, k = idx % DP;
    wihb16[idx] = f2bf(k < DIN ? wihb[n * DIN + k] : 0.0f); return; }
  idx -= NW;
  if (idx < NH) { whhf16[idx] = f2bf(whhf[idx]); return; }
  idx -= NH;
  if (idx < NH) { whhb16[idx] = f2bf(whhb[idx]); return; }
  idx -= NH;
  if (idx < G4H) { biasf[idx] = bihf[idx] + bhhf[idx]; return; }
  idx -= G4H;
  if (idx < G4H) { biasb[idx] = bihb[idx] + bhhb[idx]; return; }
}

// ---------------------------------------------------------------------------
// Embedding gather -> bf16 emb (MROWS x DP), char/pad region zeroed
// ---------------------------------------------------------------------------
__global__ __launch_bounds__(128) void build_emb(
    const int* __restrict__ x, const float* __restrict__ word_vec,
    const float* __restrict__ edge_vec, __bf16* __restrict__ emb)
{
  int row = blockIdx.x;            // t*NB + b
  int t = row / NB, b = row % NB;
  int w1 = x[b * 210 + t];
  int w2 = x[b * 210 + MAXLEN + t];
  int ed = x[b * 210 + 2 * MAXLEN + t];
  size_t base = (size_t)row * DP;
  for (int e = threadIdx.x; e < DP; e += 128) {
    float v;
    if      (e < WD)        v = word_vec[(size_t)w1 * WD + e];
    else if (e < 2 * WD)    v = word_vec[(size_t)w2 * WD + (e - WD)];
    else if (e < 2 * WD + EDIM) v = edge_vec[ed * EDIM + (e - 2 * WD)];
    else v = 0.0f;                          // char feats (filled later) + pad
    emb[base + e] = f2bf(v);
  }
}

// ---------------------------------------------------------------------------
// Char-CNN: per-batch block; stage x1 (70x300) bf16 in LDS; relu+maxpool
// ---------------------------------------------------------------------------
__global__ __launch_bounds__(128) void conv_char(
    __bf16* emb, const float* __restrict__ conv_w, const float* __restrict__ conv_b)
{
  __shared__ __bf16 sx[MAXLEN * WD];       // 42 KB
  int b = blockIdx.x;
  for (int idx = threadIdx.x; idx < MAXLEN * WD; idx += 128) {
    int t = idx / WD, e = idx % WD;
    sx[idx] = emb[((size_t)t * NB + b) * DP + e];
  }
  __syncthreads();
  int f = threadIdx.x;
  if (f < FN) {
    float maxv = 0.0f;                     // relu outputs are >= 0
    const float* wrow = conv_w + (size_t)f * 3 * WD;
    for (int p = 0; p < MAXLEN - 2; p++) {
      float s = conv_b[f];
      for (int d = 0; d < 3; d++) {
        const __bf16* xr = &sx[(p + d) * WD];
        const float* wr = wrow + d * WD;
        for (int w = 0; w < WD; w++) s += wr[w] * bf2f(xr[w]);
      }
      s = s > 0.0f ? s : 0.0f;
      maxv = fmaxf(maxv, s);
    }
    __bf16 hv = f2bf(maxv);
    for (int t = 0; t < MAXLEN; t++)
      emb[((size_t)t * NB + b) * DP + 2 * WD + EDIM + f] = hv;
  }
}

// ---------------------------------------------------------------------------
// Tiled WMMA GEMM: out[m][n] = sum_k A[m][k]*W[n][k] (+bias[n]) (+addmat[m][n])
// Block: 256 threads = 8 waves, tile M=64 x N=128, K-step 32.
// Fragment layouts follow CDNA5 ISA 7.12.2 (bf16 A 16x32, B 32x16, f32 C 16x16).
// ---------------------------------------------------------------------------
__global__ __launch_bounds__(256) void wmma_gemm_bf16(
    const __bf16* __restrict__ A, int lda,
    const __bf16* __restrict__ W, int ldw,
    const float* __restrict__ bias,
    const float* __restrict__ addmat,
    float* __restrict__ out, int ldc, int K)
{
  __shared__ __bf16 As[64 * 32];           // 4 KB
  __shared__ __bf16 Bs[128 * 32];          // 8 KB
  const int tid = threadIdx.x;
  const int lane = tid & 31, wv = tid >> 5;
  const int wm = wv & 1, wn = wv >> 1;     // 2x4 wave grid -> 64x128 tile
  const int m0 = blockIdx.y * 64;
  const int n0 = blockIdx.x * 128;

  f32x8 acc[2][2];
  const f32x8 z = {0.f,0.f,0.f,0.f,0.f,0.f,0.f,0.f};
  acc[0][0] = z; acc[0][1] = z; acc[1][0] = z; acc[1][1] = z;

  const int arow = tid >> 2, acol = (tid & 3) << 3;   // 8 bf16 per thread
  const int brow = tid >> 1, bcol = (tid & 1) << 4;   // 16 bf16 per thread
  const int half = lane >> 4, l16 = lane & 15;

  for (int k0 = 0; k0 < K; k0 += 32) {
    // stage A tile (64x32) and B tile (128x32) in LDS, 16B vector loads
    *(uint4*)(&As[arow * 32 + acol]) =
        *(const uint4*)(A + (size_t)(m0 + arow) * lda + k0 + acol);
    { const uint4* src = (const uint4*)(W + (size_t)(n0 + brow) * ldw + k0 + bcol);
      uint4* dst = (uint4*)(&Bs[brow * 32 + bcol]);
      dst[0] = src[0]; dst[1] = src[1]; }
    if (k0 + 32 < K) {
      __builtin_prefetch(A + (size_t)(m0 + arow) * lda + k0 + 32 + acol, 0, 1);
      __builtin_prefetch(W + (size_t)(n0 + brow) * ldw + k0 + 32 + bcol, 0, 1);
    }
    __syncthreads();

    bf16x16 afr[2], bfr[2];
#pragma unroll
    for (int mi = 0; mi < 2; mi++) {
      int m = wm * 32 + mi * 16 + l16;
      bf16x8 lo = *(const bf16x8*)(&As[m * 32 + 8 * half]);        // K = 8h..8h+7
      bf16x8 hi = *(const bf16x8*)(&As[m * 32 + 16 + 8 * half]);   // K = 16+8h..
      afr[mi] = SHUF16(lo, hi);
    }
#pragma unroll
    for (int ni = 0; ni < 2; ni++) {
      int n = wn * 32 + ni * 16 + l16;
      bf16x8 lo = *(const bf16x8*)(&Bs[n * 32 + 8 * half]);        // K = 8h..8h+7
      bf16x8 hi = *(const bf16x8*)(&Bs[n * 32 + 16 + 8 * half]);   // K = 16+8h..
      bfr[ni] = SHUF16(lo, hi);
    }
#pragma unroll
    for (int mi = 0; mi < 2; mi++)
#pragma unroll
      for (int ni = 0; ni < 2; ni++)
        acc[mi][ni] = __builtin_amdgcn_wmma_f32_16x16x32_bf16(
            false, afr[mi], false, bfr[ni], (short)0, acc[mi][ni], false, false);
    __syncthreads();
  }

  // epilogue: f32 C/D layout — VGPR r: M = r + 8*(lane/16), N = lane%16
#pragma unroll
  for (int mi = 0; mi < 2; mi++)
#pragma unroll
    for (int ni = 0; ni < 2; ni++) {
      int col = n0 + wn * 32 + ni * 16 + l16;
      float bval = bias ? bias[col] : 0.0f;
#pragma unroll
      for (int r = 0; r < 8; r++) {
        int row = m0 + wm * 32 + mi * 16 + 8 * half + r;
        float v = acc[mi][ni][r] + bval;
        if (addmat) v += addmat[(size_t)row * ldc + col];
        out[(size_t)row * ldc + col] = v;
      }
    }
}

// ---------------------------------------------------------------------------
// LSTM state update (gates), h also emitted as bf16 for next recurrent GEMM
// ---------------------------------------------------------------------------
__global__ __launch_bounds__(256) void zero_state(__bf16* h16, float* c)
{
  int idx = blockIdx.x * 256 + threadIdx.x;
  if (idx < NB * HDIM) { h16[idx] = f2bf(0.0f); c[idx] = 0.0f; }
}

__global__ __launch_bounds__(256) void lstm_gates(
    const float* __restrict__ g, float* __restrict__ c,
    __bf16* __restrict__ h16, float* __restrict__ hs_t)
{
  int idx = blockIdx.x * 256 + threadIdx.x;   // b*HDIM + j
  int b = idx >> 8, j = idx & 255;
  const float* gb = g + (size_t)b * G4H;
  float ig = gb[j], fg = gb[HDIM + j], gg = gb[2 * HDIM + j], og = gb[3 * HDIM + j];
  float cn = sigm(fg) * c[idx] + sigm(ig) * tanhf(gg);
  float hn = sigm(og) * tanhf(cn);
  c[idx] = cn;
  h16[idx] = f2bf(hn);
  hs_t[idx] = hn;
}

// ---------------------------------------------------------------------------
// Emission projection: em[t,b,tag] = [hf|hb] . proj_w[tag] + proj_b[tag]
// ---------------------------------------------------------------------------
__global__ __launch_bounds__(256) void project(
    const float* __restrict__ hsf, const float* __restrict__ hsb,
    const float* __restrict__ proj_w, const float* __restrict__ proj_b,
    float* __restrict__ em)
{
  int idx = blockIdx.x * 256 + threadIdx.x;   // t*NB + b
  if (idx >= MROWS) return;
  const float* hf = hsf + (size_t)idx * HDIM;
  const float* hb = hsb + (size_t)idx * HDIM;
  for (int tag = 0; tag < NT; tag++) {
    const float* wr = proj_w + (size_t)tag * 2 * HDIM;
    float s = proj_b[tag];
    for (int j = 0; j < HDIM; j++) s += hf[j] * wr[j];
    for (int j = 0; j < HDIM; j++) s += hb[j] * wr[HDIM + j];
    em[(size_t)idx * NT + tag] = s;
  }
}

// ---------------------------------------------------------------------------
// Per-batch Viterbi decode (T=9, first-max argmax like jnp.argmax)
// ---------------------------------------------------------------------------
__global__ __launch_bounds__(256) void viterbi(
    const float* __restrict__ em, const float* __restrict__ start_t,
    const float* __restrict__ end_t, const float* __restrict__ trans,
    unsigned char* __restrict__ bp, float* __restrict__ out)
{
  int b = blockIdx.x * 256 + threadIdx.x;
  if (b >= NB) return;
  float score[NT], nscore[NT];
  for (int j = 0; j < NT; j++) score[j] = start_t[j] + em[(size_t)b * NT + j];
  for (int t = 1; t < MAXLEN; t++) {
    const float* emt = em + ((size_t)t * NB + b) * NT;
    unsigned char* bpt = bp + ((size_t)b * MAXLEN + t) * NT;
    for (int j = 0; j < NT; j++) {
      float best = score[0] + trans[j];
      int bi = 0;
      for (int i = 1; i < NT; i++) {
        float cnd = score[i] + trans[i * NT + j];
        if (cnd > best) { best = cnd; bi = i; }
      }
      nscore[j] = best + emt[j];
      bpt[j] = (unsigned char)bi;
    }
    for (int j = 0; j < NT; j++) score[j] = nscore[j];
  }
  float best = score[0] + end_t[0];
  int tag = 0;
  for (int j = 1; j < NT; j++) {
    float s = score[j] + end_t[j];
    if (s > best) { best = s; tag = j; }
  }
  out[(size_t)b * MAXLEN + (MAXLEN - 1)] = (float)tag;
  for (int t = MAXLEN - 1; t >= 1; t--) {
    tag = bp[((size_t)b * MAXLEN + t) * NT + tag];
    out[(size_t)b * MAXLEN + (t - 1)] = (float)tag;
  }
}

// ---------------------------------------------------------------------------
extern "C" void kernel_launch(void* const* d_in, const int* in_sizes, int n_in,
                              void* d_out, int out_size, void* d_ws, size_t ws_size,
                              hipStream_t stream)
{
  (void)in_sizes; (void)n_in; (void)out_size; (void)ws_size;
  const int*   x        = (const int*)  d_in[0];
  const float* word_vec = (const float*)d_in[1];
  const float* edge_vec = (const float*)d_in[2];
  const float* conv_w   = (const float*)d_in[3];
  const float* conv_b   = (const float*)d_in[4];
  const float* wih_f    = (const float*)d_in[5];
  const float* whh_f    = (const float*)d_in[6];
  const float* bih_f    = (const float*)d_in[7];
  const float* bhh_f    = (const float*)d_in[8];
  const float* wih_b    = (const float*)d_in[9];
  const float* whh_b    = (const float*)d_in[10];
  const float* bih_b    = (const float*)d_in[11];
  const float* bhh_b    = (const float*)d_in[12];
  const float* proj_w   = (const float*)d_in[13];
  const float* proj_b   = (const float*)d_in[14];
  const float* start_t  = (const float*)d_in[15];
  const float* end_t    = (const float*)d_in[16];
  const float* trans    = (const float*)d_in[17];
  float* out = (float*)d_out;

  char* ws = (char*)d_ws;
  size_t off = 0;
  auto take = [&](size_t bytes) -> char* {
    char* p = ws + off;
    off += (bytes + 255) & ~(size_t)255;
    return p;
  };
  __bf16* emb16   = (__bf16*)take((size_t)MROWS * DP * 2);
  __bf16* wihf16  = (__bf16*)take((size_t)G4H * DP * 2);
  __bf16* wihb16  = (__bf16*)take((size_t)G4H * DP * 2);
  __bf16* whhf16  = (__bf16*)take((size_t)G4H * HDIM * 2);
  __bf16* whhb16  = (__bf16*)take((size_t)G4H * HDIM * 2);
  float*  biasf   = (float*) take((size_t)G4H * 4);
  float*  biasb   = (float*) take((size_t)G4H * 4);
  float*  pre     = (float*) take((size_t)MROWS * G4H * 4);   // reused per dir
  float*  gbuf    = (float*) take((size_t)NB * G4H * 4);
  __bf16* h16     = (__bf16*)take((size_t)NB * HDIM * 2);
  float*  cst     = (float*) take((size_t)NB * HDIM * 4);
  float*  hsf     = (float*) take((size_t)MROWS * HDIM * 4);
  float*  hsb     = (float*) take((size_t)MROWS * HDIM * 4);
  float*  em      = (float*) take((size_t)MROWS * NT * 4);
  unsigned char* bp = (unsigned char*)take((size_t)NB * MAXLEN * NT);

  // 1) weights -> bf16 (padded), combined biases
  {
    long total = 2L * G4H * DP + 2L * G4H * HDIM + 2L * G4H;
    int blocks = (int)((total + 255) / 256);
    convert_weights<<<blocks, 256, 0, stream>>>(
        wih_f, wih_b, whh_f, whh_b, bih_f, bhh_f, bih_b, bhh_b,
        wihf16, wihb16, whhf16, whhb16, biasf, biasb);
  }
  // 2) embeddings, 3) char-CNN features into emb
  build_emb<<<MROWS, 128, 0, stream>>>(x, word_vec, edge_vec, emb16);
  conv_char<<<NB, 128, 0, stream>>>(emb16, conv_w, conv_b);

  // 4) BiLSTM, one direction at a time (pre buffer reused)
  for (int dir = 0; dir < 2; dir++) {
    const __bf16* wih16 = dir ? wihb16 : wihf16;
    const __bf16* whh16 = dir ? whhb16 : whhf16;
    const float*  bias  = dir ? biasb  : biasf;
    float* hs = dir ? hsb : hsf;

    // time-parallel input projection: (35840 x 736) @ (736 -> 1024) + bias
    wmma_gemm_bf16<<<dim3(G4H / 128, MROWS / 64), 256, 0, stream>>>(
        emb16, DP, wih16, DP, bias, nullptr, pre, G4H, DP);

    zero_state<<<(NB * HDIM + 255) / 256, 256, 0, stream>>>(h16, cst);

    for (int s = 0; s < MAXLEN; s++) {
      int t = dir ? (MAXLEN - 1 - s) : s;
      // g = pre[t] + h @ Whh^T : (512 x 256) @ (256 -> 1024)
      wmma_gemm_bf16<<<dim3(G4H / 128, NB / 64), 256, 0, stream>>>(
          h16, HDIM, whh16, HDIM, nullptr, pre + (size_t)t * NB * G4H,
          gbuf, G4H, HDIM);
      lstm_gates<<<(NB * HDIM) / 256, 256, 0, stream>>>(
          gbuf, cst, h16, hs + (size_t)t * NB * HDIM);
    }
  }

  // 5) projection + 6) Viterbi
  project<<<(MROWS + 255) / 256, 256, 0, stream>>>(hsf, hsb, proj_w, proj_b, em);
  viterbi<<<(NB + 255) / 256, 256, 0, stream>>>(em, start_t, end_t, trans, bp, out);
}
